// ParallelRetention_32727650795913
// MI455X (gfx1250) — compile-verified, matrix operand order checked by reading, not run on hardware
//
#include <hip/hip_runtime.h>

typedef __attribute__((ext_vector_type(16))) _Float16 v16h;
typedef __attribute__((ext_vector_type(8)))  _Float16 v8h;
typedef __attribute__((ext_vector_type(8)))  float    v8f;

#define B_ 4
#define S_ 4096
#define F_ 128
// log2(0.96875)
#define L2GAMMA (-0.04580369f)

union Frag16 { v16h v; v8h h[2]; };

// Loads a 16x32 f16 fragment in the CDNA5 WMMA A-operand layout from a
// row-major matrix (also valid for B operands: B is column-striped, so
// loading the *rows* of the transposed-source with this loader produces B).
// Lane layout (ISA 7.12.2): lanes 0-15 = rows 0-15, K in {0..7,16..23};
// lanes 16-31 = rows 0-15, K in {8..15,24..31}.
__device__ __forceinline__ v16h load_frag(const _Float16* base, int row, int ld,
                                          int kbase, int lane) {
  int hs = (lane & 16) ? 8 : 0;
  const _Float16* p = base + (size_t)row * ld + kbase + hs;
  Frag16 f;
  f.h[0] = *(const v8h*)(p);        // elements 0..7  -> K = kbase+hs .. +7
  f.h[1] = *(const v8h*)(p + 16);   // elements 8..15 -> K = kbase+hs+16 .. +23
  return f.v;
}

__device__ __forceinline__ v8f wmma_f16(v16h a, v16h b, v8f c) {
  return __builtin_amdgcn_wmma_f32_16x16x32_f16(false, a, false, b,
                                                (short)0, c, false, false);
}

// ---------------------------------------------------------------- prep ----
__global__ void prep_kernel(const float* __restrict__ x,
                            const float* __restrict__ Wq,
                            const float* __restrict__ Wk,
                            const float* __restrict__ Wv,
                            _Float16* __restrict__ Xh,
                            _Float16* __restrict__ Whq,
                            _Float16* __restrict__ Whk,
                            _Float16* __restrict__ Whv) {
  int i = blockIdx.x * blockDim.x + threadIdx.x;
  if (i < B_ * S_ * F_) Xh[i] = (_Float16)x[i];
  if (i < F_ * F_) {
    Whq[i] = (_Float16)Wq[i];
    Whk[i] = (_Float16)Wk[i];
    Whv[i] = (_Float16)Wv[i];
  }
}

// ---------------------------------------------------------------- proj ----
// One wave per 16 token rows. Emits Qh/Kh = [q*cos | q*sin] f16 [B*S,256]
// and V transposed f16 [B,F,S]. cos/sin for the 8 rows of a column come from
// two precise sincosf calls + a Givens-rotation recurrence (error ~1e-6).
__global__ __launch_bounds__(256) void proj_kernel(
    const _Float16* __restrict__ Xh,
    const _Float16* __restrict__ Whq, const _Float16* __restrict__ Whk,
    const _Float16* __restrict__ Whv,
    const float* __restrict__ bq, const float* __restrict__ bk,
    const float* __restrict__ bv, const float* __restrict__ theta,
    _Float16* __restrict__ Qh, _Float16* __restrict__ Kh,
    _Float16* __restrict__ Vt) {
  int lane = threadIdx.x & 31;
  int w    = (blockIdx.x * blockDim.x + threadIdx.x) >> 5;
  int row0 = w * 16;                 // global token row, [0, B*S)
  int b    = row0 / S_;
  int sp0  = row0 - b * S_;          // sequence position of tile row 0
  int rl   = lane & 15;
  int hs8  = (lane & 16) ? 8 : 0;

  v16h xf[4];
#pragma unroll
  for (int kk = 0; kk < 4; ++kk)
    xf[kk] = load_frag(Xh, row0 + rl, F_, kk * 32, lane);

#pragma unroll 1
  for (int o0 = 0; o0 < F_; o0 += 16) {
    int o = o0 + rl;
    v8f aq = {}, ak = {}, av = {};
#pragma unroll
    for (int kk = 0; kk < 4; ++kk) {
      v16h bqf = load_frag(Whq, o0 + rl, F_, kk * 32, lane);
      aq = wmma_f16(xf[kk], bqf, aq);
      v16h bkf = load_frag(Whk, o0 + rl, F_, kk * 32, lane);
      ak = wmma_f16(xf[kk], bkf, ak);
      v16h bvf = load_frag(Whv, o0 + rl, F_, kk * 32, lane);
      av = wmma_f16(xf[kk], bvf, av);
    }
    float biq = bq[o], bik = bk[o], biv = bv[o];
    float th  = theta[o];
    // precise base angle + step, then rotate (2 FMAs) for rows 1..7
    float ph0 = (float)(sp0 + hs8 + 1) * th;
    float c, sn, cd, sd;
    sincosf(ph0, &sn, &c);
    sincosf(th, &sd, &cd);
    v8h vvec;
#pragma unroll
    for (int r = 0; r < 8; ++r) {
      int   sp = sp0 + r + hs8;      // C-layout: row m = r + 8*(lane>=16)
      float qv = aq[r] + biq;
      float kv = ak[r] + bik;
      float vv = av[r] + biv;
      size_t ro = ((size_t)b * S_ + sp) * 256;
      Qh[ro + o]      = (_Float16)(qv * c);
      Qh[ro + F_ + o] = (_Float16)(qv * sn);
      Kh[ro + o]      = (_Float16)(kv * c);
      Kh[ro + F_ + o] = (_Float16)(kv * sn);
      vvec[r] = (_Float16)vv;
      float cn  = c * cd - sn * sd;  // advance phase by theta
      float snn = sn * cd + c * sd;
      c = cn; sn = snn;
    }
    // Vt[b][o][sp0+hs8 .. +7] : contiguous, 16B aligned
    *(v8h*)(Vt + ((size_t)(b * F_ + o)) * S_ + sp0 + hs8) = vvec;
  }
}

// ---------------------------------------------------------------- main ----
// One 128-thread block (4 waves) per 16-query block. Wave p handles key
// blocks p*32, p*32+128, ... (4-way split-K over the causal range), then the
// four partial 16x128 f32 accumulators are summed through LDS in a fixed
// order (deterministic) and wave 0 stores.
__global__ __launch_bounds__(128) void retention_kernel(
    const _Float16* __restrict__ Qh, const _Float16* __restrict__ Kh,
    const _Float16* __restrict__ Vt, float* __restrict__ out) {
  __shared__ _Float16 sP[4 * 16 * 32];       // per-wave score staging (4 KB)
  __shared__ float    sRed[3 * 8 * 8 * 32];  // partials of waves 1..3 (24 KB)
  int lane = threadIdx.x & 31;
  int wloc = threadIdx.x >> 5;             // 0..3
  int b    = blockIdx.x >> 8;              // 256 row-blocks per batch
  int i0   = (blockIdx.x & 255) << 4;
  int rl   = lane & 15;
  int hs8  = (lane & 16) ? 8 : 0;

  const _Float16* Qb = Qh + (size_t)b * S_ * 256;
  const _Float16* Kb = Kh + (size_t)b * S_ * 256;
  const _Float16* Vb = Vt + (size_t)b * F_ * S_;
  _Float16* myP = sP + wloc * (16 * 32);

  v16h qf[8];
#pragma unroll
  for (int f = 0; f < 8; ++f)
    qf[f] = load_frag(Qb, i0 + rl, 256, f * 32, lane);

  v8f acc[8] = {};

  for (int j0 = wloc * 32; j0 <= i0; j0 += 128) {
    v8f s0 = {}, s1 = {};
#pragma unroll
    for (int f = 0; f < 8; ++f) {
      v16h k0 = load_frag(Kb, j0 + rl, 256, f * 32, lane);
      s0 = wmma_f16(qf[f], k0, s0);
      v16h k1 = load_frag(Kb, j0 + 16 + rl, 256, f * 32, lane);
      s1 = wmma_f16(qf[f], k1, s1);
    }
    // decay (gamma^(i-j), 0 for j>i): single v_exp_f32 per element (TRANS,
    // co-executes with WMMA); args are in [-188,0], underflow == reference
#pragma unroll
    for (int r = 0; r < 8; ++r) {
      int   m  = r + hs8;
      int   qi = i0 + m;
      int   d0 = qi - (j0 + rl);
      int   d1 = d0 - 16;
      float w0 = (d0 >= 0) ? __builtin_amdgcn_exp2f(L2GAMMA * (float)d0) : 0.0f;
      float w1 = (d1 >= 0) ? __builtin_amdgcn_exp2f(L2GAMMA * (float)d1) : 0.0f;
      myP[m * 32 + rl]      = (_Float16)(s0[r] * w0);
      myP[m * 32 + 16 + rl] = (_Float16)(s1[r] * w1);
    }
    // reload as A-fragment (same-wave DS ops are in-order)
    v16h pf = load_frag(myP, rl, 32, 0, lane);
#pragma unroll
    for (int fi = 0; fi < 8; ++fi) {
      v16h vf = load_frag(Vb, fi * 16 + rl, S_, j0, lane);
      acc[fi] = wmma_f16(pf, vf, acc[fi]);
    }
  }

  // deterministic cross-wave reduction of the 16x128 f32 tile
  if (wloc > 0) {
    float* dst = sRed + (wloc - 1) * 2048;
#pragma unroll
    for (int fi = 0; fi < 8; ++fi)
#pragma unroll
      for (int r = 0; r < 8; ++r)
        dst[fi * 256 + r * 32 + lane] = acc[fi][r];
  }
  __syncthreads();
  if (wloc == 0) {
    float* ob = out + (size_t)b * S_ * F_;
#pragma unroll
    for (int fi = 0; fi < 8; ++fi) {
#pragma unroll
      for (int r = 0; r < 8; ++r) {
        int   idx = fi * 256 + r * 32 + lane;
        float s   = acc[fi][r];
        s += sRed[idx];
        s += sRed[2048 + idx];
        s += sRed[4096 + idx];
        ob[(size_t)(i0 + r + hs8) * F_ + fi * 16 + rl] = s;
      }
    }
  }
}

// ---------------------------------------------------------------- launch --
extern "C" void kernel_launch(void* const* d_in, const int* in_sizes, int n_in,
                              void* d_out, int out_size, void* d_ws,
                              size_t ws_size, hipStream_t stream) {
  const float* x     = (const float*)d_in[0];
  const float* Wq    = (const float*)d_in[1];
  const float* bq    = (const float*)d_in[2];
  const float* Wk    = (const float*)d_in[3];
  const float* bk    = (const float*)d_in[4];
  const float* Wv    = (const float*)d_in[5];
  const float* bv    = (const float*)d_in[6];
  const float* theta = (const float*)d_in[7];
  float* out = (float*)d_out;

  char*  ws  = (char*)d_ws;
  size_t off = 0;
  auto alloc = [&](size_t bytes) {
    char* p = ws + off;
    off += (bytes + 255) & ~(size_t)255;
    return p;
  };
  _Float16* Xh  = (_Float16*)alloc((size_t)B_ * S_ * F_ * 2);
  _Float16* Whq = (_Float16*)alloc((size_t)F_ * F_ * 2);
  _Float16* Whk = (_Float16*)alloc((size_t)F_ * F_ * 2);
  _Float16* Whv = (_Float16*)alloc((size_t)F_ * F_ * 2);
  _Float16* Qh  = (_Float16*)alloc((size_t)B_ * S_ * 256 * 2);
  _Float16* Kh  = (_Float16*)alloc((size_t)B_ * S_ * 256 * 2);
  _Float16* Vt  = (_Float16*)alloc((size_t)B_ * F_ * S_ * 2);

  int nprep = B_ * S_ * F_;
  prep_kernel<<<(nprep + 255) / 256, 256, 0, stream>>>(x, Wq, Wk, Wv, Xh, Whq,
                                                       Whk, Whv);
  int nblk = (B_ * S_ / 16) * 32 / 256;  // 1024 waves -> 128 blocks
  proj_kernel<<<nblk, 256, 0, stream>>>(Xh, Whq, Whk, Whv, bq, bk, bv, theta,
                                        Qh, Kh, Vt);
  // one 4-wave block per 16-row query block: B * (S/16) = 1024 blocks
  retention_kernel<<<B_ * (S_ / 16), 128, 0, stream>>>(Qh, Kh, Vt, out);
}
